// AASISTInference_7395933683796
// MI455X (gfx1250) — matrix operand order repaired and checked by reading, not compiled, not run on hardware
//
#include <hip/hip_runtime.h>

// ============================================================================
// AASIST inference for gfx1250 (MI455X).  Compile-only target.
//
// Strategy:
//  * All encoder convs with cin>=32 run as fp16 WMMA GEMMs
//    (v_wmma_f32_16x16x32_f16), activations kept NHWC f16.
//  * Sinc conv fused with |.|, maxpool(3,3), first_bn, selu (WMMA, K=128).
//  * conv1 fused with bias+bn2+selu epilogue; conv2 fused with identity/down
//    add + maxpool(1,3) epilogue (down conv folded in as 2nd WMMA K-loop).
//  * global_prefetch_b8 (via __builtin_prefetch) streams the next W-tile
//    while the current tile's WMMAs run.
//  * Tiny graph-attention stages: one fused workgroup per batch item (VALU).
//
// d_in leaf order assumption: jax.tree_util flatten (sorted dict keys):
//  enc0{bn2.b,bn2.s,c1.b,c1.w,c2.b,c2.w,dn.b,dn.w}=0..7
//  enc1{bn1.b,bn1.s,bn2.b,bn2.s,c1.b,c1.w,c2.b,c2.w}=8..15
//  enc2{bn1.b,bn1.s,bn2.b,bn2.s,c1.b,c1.w,c2.b,c2.w,dn.b,dn.w}=16..25
//  enc3=26..33 enc4=34..41 enc5=42..49
//  first_bn.b=50 first_bn.s=51
//  gat_S{ap.b,ap.w,att_w,bn.b,bn.s,po.b,po.w,pw.b,pw.w}=52..60  gat_T=61..69
//  master1=70 master2=71 out.b=72 out.w=73
//  pool_S{b,w}=74,75 pool_T=76,77 pool_hS1=78,79 pool_hS2=80,81
//  pool_hT1=82,83 pool_hT2=84,85  pos_S=86 sinc=87
//  st11=88..109 st12=110..131 st21=132..153 st22=154..175 (22 leaves each:
//   ap.b,ap.w,apM.b,apM.w,bn.b,bn.s,po.b,po.w,poM.b,poM.w,pt1.b,pt1.w,
//   pt2.b,pt2.w,pw.b,pw.w,pwM.b,pwM.w,w11,w12,w22,wM)
//  x=176
// ============================================================================

typedef _Float16 f16;
typedef _Float16 v16h __attribute__((ext_vector_type(16)));
typedef _Float16 v8h  __attribute__((ext_vector_type(8)));
typedef float    v8f  __attribute__((ext_vector_type(8)));

#define WMMA(a,b,c) __builtin_amdgcn_wmma_f32_16x16x32_f16(false,(a),false,(b),(short)0,(c),false,false)

__device__ __forceinline__ float selu_f(float x){
  const float l=1.0507009873554805f, a=1.6732632423543772f;
  return x>0.f ? l*x : l*a*(expf(x)-1.f);
}

// A-matrix lane fragment: lanes 0-15 hold K {0..7,16..23}; lanes 16-31 hold
// K {8..15,24..31} of the 32-wide k-step (ISA 7.12.2, 16-bit A 16x32).
__device__ __forceinline__ v16h ldA(const f16* p, int lh){
  v8h lo, hi;
  __builtin_memcpy(&lo, p + 8*lh, 16);
  __builtin_memcpy(&hi, p + 16 + 8*lh, 16);
  return __builtin_shufflevector(lo, hi, 0,1,2,3,4,5,6,7,8,9,10,11,12,13,14,15);
}
// B-matrix lane fragment: lanes 0-15 hold K {0..7,16..23}, lanes 16-31 hold
// K {8..15,24..31} of column N=lane%16 (same order as ldA).
__device__ __forceinline__ v16h ldB(const f16* p, int lh){
  v8h lo, hi;
  __builtin_memcpy(&lo, p + 8*lh, 16);
  __builtin_memcpy(&hi, p + 16 + 8*lh, 16);
  return __builtin_shufflevector(lo, hi, 0,1,2,3,4,5,6,7,8,9,10,11,12,13,14,15);
}

// ---------------------------------------------------------------------------
// small utility kernels
// ---------------------------------------------------------------------------
__global__ void k_cvt(const float* __restrict__ a, f16* __restrict__ o, int n){
  int i = blockIdx.x*256 + threadIdx.x;
  if(i<n) o[i] = (f16)a[i];
}

// pack OIHW f32 -> f16 WT[co][k], k=(kh*KW+kw)*CI+ci, zero-padded to COp x Kp
__global__ void k_pack_conv(const float* __restrict__ w, f16* __restrict__ o,
                            int CO,int CI,int KH,int KW,int COp,int Kp){
  int i = blockIdx.x*256 + threadIdx.x;
  int tot = COp*Kp; if(i>=tot) return;
  int co = i / Kp, k = i % Kp;
  int K = KH*KW*CI;
  f16 v = (f16)0.f;
  if(co<CO && k<K){
    int khkw = k / CI, ci = k % CI;
    int kh = khkw / KW, kw = khkw % KW;
    v = (f16)w[((co*CI+ci)*KH+kh)*KW+kw];
  }
  o[i] = v;
}

__global__ void k_bnselu(const f16* __restrict__ a, f16* __restrict__ o,
                         const float* __restrict__ s, const float* __restrict__ bb,
                         long n, int cmask){
  long i = (long)blockIdx.x*256 + threadIdx.x;
  if(i>=n) return;
  int c = (int)(i & cmask);
  o[i] = (f16)selu_f((float)a[i]*s[c]+bb[c]);
}

__global__ void k_add(float* a, const float* b, int n){
  int i = blockIdx.x*256+threadIdx.x; if(i<n) a[i]+=b[i];
}
__global__ void k_maxel(const float* a, const float* b, float* o, int n){
  int i = blockIdx.x*256+threadIdx.x; if(i<n) o[i]=fmaxf(a[i],b[i]);
}
__global__ void k_bcast(const float* src, float* dst, int D){
  int b=blockIdx.x;
  for(int d=threadIdx.x; d<D; d+=blockDim.x) dst[b*D+d]=src[d];
}

// ---------------------------------------------------------------------------
// Stage 1: sinc conv (GEMM K=128, N=80 padded filters) fused with
// |.| + maxpool(3,3) + first_bn + selu.  One wave -> 48 conv positions x 80
// filters -> 16 pooled positions x 23 filter-groups.
// ---------------------------------------------------------------------------
#define NB_W0 26624
__global__ __launch_bounds__(32) void k_sincconv(
    const f16* __restrict__ xh, const f16* __restrict__ wp,
    const float* __restrict__ s0p, const float* __restrict__ b0p,
    f16* __restrict__ act0){
  __shared__ float cb[48*80];
  int tile = blockIdx.x, b = blockIdx.y;
  int lane = threadIdx.x, lm = lane&15, lh = lane>>4;
  int p0 = 48*tile;
  const f16* xb = xh + (size_t)b*80000;
  // stream samples of tile+32 into cache while this tile computes
  if(p0 + 48*32 + lane*8 + 174 < 80000)
    __builtin_prefetch(xb + p0 + 48*32 + lane*8, 0, 1);
  v8f acc[15]; v8f zf = {};
  #pragma unroll
  for(int i=0;i<15;++i) acc[i]=zf;
  #pragma unroll
  for(int ks=0; ks<4; ++ks){
    v16h a0 = ldA(xb + p0 + 0  + lm + 32*ks, lh);
    v16h a1 = ldA(xb + p0 + 16 + lm + 32*ks, lh);
    v16h a2 = ldA(xb + p0 + 32 + lm + 32*ks, lh);
    #pragma unroll
    for(int nt=0; nt<5; ++nt){
      v16h bm = ldB(wp + (size_t)(16*nt+lm)*128 + 32*ks, lh);
      acc[nt*3+0] = WMMA(a0, bm, acc[nt*3+0]);
      acc[nt*3+1] = WMMA(a1, bm, acc[nt*3+1]);
      acc[nt*3+2] = WMMA(a2, bm, acc[nt*3+2]);
    }
  }
  #pragma unroll
  for(int nt=0; nt<5; ++nt)
    #pragma unroll
    for(int mt=0; mt<3; ++mt)
      #pragma unroll
      for(int j=0; j<8; ++j)
        cb[(16*mt + 8*lh + j)*80 + 16*nt + lm] = acc[nt*3+mt][j];
  __syncthreads();
  float s0 = s0p[0], b0 = b0p[0];
  for(int o=lane; o<368; o+=32){
    int fh = o>>4, wo = o&15;
    float m = 0.f;
    for(int t=0;t<3;++t)
      for(int fr=0; fr<3; ++fr)
        m = fmaxf(m, fabsf(cb[(3*wo+t)*80 + 3*fh+fr]));
    act0[((size_t)b*23 + fh)*NB_W0 + 16*tile + wo] = (f16)selu_f(m*s0+b0);
  }
}

// ---------------------------------------------------------------------------
// block0 conv1 (cin=1, VALU) fused with bn2+selu -> y1 (B,24,W,32) f16
// ---------------------------------------------------------------------------
__global__ void k_conv1_c1(const f16* __restrict__ act0,
                           const float* __restrict__ w, const float* __restrict__ bias,
                           const float* __restrict__ bns, const float* __restrict__ bnb,
                           f16* __restrict__ y1){
  __shared__ float wl[192], bl[32], sl[32], bbl[32];
  int t = threadIdx.x;
  if(t<192) wl[t]=w[t];
  if(t<32){ bl[t]=bias[t]; sl[t]=bns[t]; bbl[t]=bnb[t]; }
  __syncthreads();
  int wpos = blockIdx.x*256 + t; if(wpos>=NB_W0) return;
  int h = blockIdx.y, b = blockIdx.z;
  float in[2][3];
  for(int kh=0; kh<2; ++kh){
    int hin = h-1+kh;
    for(int kw=0; kw<3; ++kw){
      int wi = wpos-1+kw;
      in[kh][kw] = (hin>=0 && hin<23 && wi>=0 && wi<NB_W0)
                 ? (float)act0[((size_t)b*23+hin)*NB_W0+wi] : 0.f;
    }
  }
  f16* orow = y1 + (((size_t)(b*24+h))*NB_W0 + wpos)*32;
  for(int co=0; co<32; ++co){
    float a = bl[co];
    #pragma unroll
    for(int kh=0;kh<2;++kh)
      #pragma unroll
      for(int kw=0;kw<3;++kw)
        a += in[kh][kw]*wl[co*6+kh*3+kw];
    orow[co] = (f16)selu_f(a*sl[co]+bbl[co]);
  }
}

// ---------------------------------------------------------------------------
// conv1 (generic, cin>=32): WMMA GEMM, K=6C, epilogue bias+bn2+selu -> f16
// in: (B,23,W,C) f16   out: (B,24,W,CO) f16
// ---------------------------------------------------------------------------
__global__ __launch_bounds__(32) void k_convA(
    const f16* __restrict__ in, const f16* __restrict__ wt,
    const float* __restrict__ bias, const float* __restrict__ bns,
    const float* __restrict__ bnb, f16* __restrict__ out,
    int W, int C, int CO){
  int lane=threadIdx.x, lm=lane&15, lh=lane>>4;
  int w0 = 16*blockIdx.x, h = blockIdx.y;
  int cot = blockIdx.z % (CO>>4), b = blockIdx.z / (CO>>4);
  int co = 16*cot + lm;
  int K = 6*C;
  // prefetch next W-tile of the (clamped) input row -> global_prefetch_b8
  {
    int hp = h<23 ? h : 22;
    int wpf = w0 + 16 + lm; if(wpf >= W) wpf = W-1;
    __builtin_prefetch(in + (size_t)(b*23+hp)*W*C + (size_t)wpf*C, 0, 1);
  }
  v8f acc = {}; v16h zer = {};
  for(int kh=0; kh<2; ++kh){
    int hin = h-1+kh;
    if(hin<0 || hin>=23) continue;
    const f16* inr = in + (size_t)(b*23+hin)*W*C;
    for(int kw=0; kw<3; ++kw){
      int wx = w0 + lm - 1 + kw;
      bool ok = (wx>=0 && wx<W);
      const f16* ap = inr + (size_t)(ok?wx:0)*C;
      const f16* bp = wt + (size_t)co*K + (kh*3+kw)*C;
      for(int cs=0; cs<C; cs+=32){
        v16h a  = ok ? ldA(ap+cs, lh) : zer;
        v16h bm = ldB(bp+cs, lh);
        acc = WMMA(a, bm, acc);
      }
    }
  }
  float bsv=bias[co], sv=bns[co], bv=bnb[co];
  #pragma unroll
  for(int j=0;j<8;++j){
    int wpos = w0 + 8*lh + j;
    if(wpos<W)
      out[((size_t)(b*24+h)*W + wpos)*CO + co] = (f16)selu_f((acc[j]+bsv)*sv+bv);
  }
}

// ---------------------------------------------------------------------------
// conv2 fused: WMMA GEMM over y1 (3 M-tiles = 48 conv positions), optional
// down-conv folded in as 2nd GEMM (mode 1) or cin==1 down in epilogue
// (mode 2) or identity=x (mode 0); epilogue: +identity, maxpool(1,3), +bias.
// y1: (B,24,W,C) f16,  xin: (B,23,W,Cx) f16,  out: (B,23,Wp,CO) f16
// ---------------------------------------------------------------------------
__global__ __launch_bounds__(32) void k_convB(
    const f16* __restrict__ y1, const f16* __restrict__ wt2,
    const float* __restrict__ bias2, const f16* __restrict__ xin,
    const f16* __restrict__ wtd, const float* __restrict__ biasd,
    const float* __restrict__ wdc1, const float* __restrict__ bdc1,
    f16* __restrict__ out, int W, int Wp, int C, int Cx, int CO, int mode){
  __shared__ float cb[48*16];
  int lane=threadIdx.x, lm=lane&15, lh=lane>>4;
  int wp0 = 16*blockIdx.x, h = blockIdx.y;
  int cot = blockIdx.z % (CO>>4), b = blockIdx.z / (CO>>4);
  int co = 16*cot + lm;
  int p0 = 3*wp0;
  int K = 6*C;
  // prefetch next pooled tile's 48 y1 positions -> global_prefetch_b8
  {
    int ppf = p0 + 48 + lane; if(ppf >= W) ppf = W-1;
    __builtin_prefetch(y1 + (size_t)(b*24+h)*W*C + (size_t)ppf*C, 0, 1);
  }
  v8f acc0={}, acc1={}, acc2={}; v16h zer = {};
  for(int kh=0; kh<2; ++kh){
    const f16* inr = y1 + (size_t)(b*24+h+kh)*W*C;
    for(int kw=0; kw<3; ++kw){
      const f16* bp = wt2 + (size_t)co*K + (kh*3+kw)*C;
      int wx0 = p0 + lm - 1 + kw;
      for(int cs=0; cs<C; cs+=32){
        v16h bm = ldB(bp+cs, lh);
        v16h a;
        a = (wx0>=0 && wx0<W) ? ldA(inr+(size_t)wx0*C+cs,lh) : zer; acc0=WMMA(a,bm,acc0);
        int wx1=wx0+16; a = (wx1<W) ? ldA(inr+(size_t)wx1*C+cs,lh) : zer; acc1=WMMA(a,bm,acc1);
        int wx2=wx0+32; a = (wx2<W) ? ldA(inr+(size_t)wx2*C+cs,lh) : zer; acc2=WMMA(a,bm,acc2);
      }
    }
  }
  if(mode==1){ // downsample conv folded in: K2 = 3*Cx
    int K2 = 3*Cx;
    const f16* xr = xin + (size_t)(b*23+h)*W*Cx;
    for(int kw=0; kw<3; ++kw){
      const f16* bp = wtd + (size_t)co*K2 + kw*Cx;
      int wx0 = p0 + lm - 1 + kw;
      for(int cs=0; cs<Cx; cs+=32){
        v16h bm = ldB(bp+cs, lh);
        v16h a;
        a = (wx0>=0 && wx0<W) ? ldA(xr+(size_t)wx0*Cx+cs,lh) : zer; acc0=WMMA(a,bm,acc0);
        int wx1=wx0+16; a = (wx1<W) ? ldA(xr+(size_t)wx1*Cx+cs,lh) : zer; acc1=WMMA(a,bm,acc1);
        int wx2=wx0+32; a = (wx2<W) ? ldA(xr+(size_t)wx2*Cx+cs,lh) : zer; acc2=WMMA(a,bm,acc2);
      }
    }
  }
  #pragma unroll
  for(int j=0;j<8;++j){
    cb[( 0 + 8*lh + j)*16 + lm] = acc0[j];
    cb[(16 + 8*lh + j)*16 + lm] = acc1[j];
    cb[(32 + 8*lh + j)*16 + lm] = acc2[j];
  }
  __syncthreads();
  for(int it=0; it<8; ++it){
    int o = lane + 32*it;
    int wo = o>>4, col = o&15;
    int wop = wp0 + wo; if(wop>=Wp) continue;
    int cog = 16*cot + col;
    float mx = -3.0e38f;
    for(int t3=0; t3<3; ++t3){
      int p = 3*wop + t3;
      float id = 0.f;
      if(mode==0) id = (float)xin[((size_t)(b*23+h)*W + p)*Cx + cog];
      else if(mode==2){
        for(int kw=0; kw<3; ++kw){
          int wx = p-1+kw;
          if(wx>=0 && wx<W) id += (float)xin[(size_t)(b*23+h)*W + wx]*wdc1[cog*3+kw];
        }
      }
      mx = fmaxf(mx, cb[(3*wo+t3)*16 + col] + id);
    }
    float vv = mx + bias2[cog];
    if(mode==1) vv += biasd[cog];
    else if(mode==2) vv += bdc1[cog];
    out[((size_t)(b*23+h)*Wp + wop)*CO + cog] = (f16)vv;
  }
}

// ---------------------------------------------------------------------------
// graph feature extraction from e = (B,23,36,64) f16
// ---------------------------------------------------------------------------
__global__ void k_eS(const f16* __restrict__ e, const float* __restrict__ posS,
                     float* __restrict__ eS){
  int s = blockIdx.x, b = blockIdx.y, d = threadIdx.x;
  float mx = 0.f;
  for(int w=0; w<36; ++w)
    mx = fmaxf(mx, fabsf((float)e[(((size_t)(b*23+s))*36 + w)*64 + d]));
  eS[((size_t)b*23+s)*64 + d] = mx + posS[s*64+d];
}
__global__ void k_eT(const f16* __restrict__ e, float* __restrict__ eT){
  int w = blockIdx.x, b = blockIdx.y, d = threadIdx.x;
  float mx = 0.f;
  for(int s=0; s<23; ++s)
    mx = fmaxf(mx, fabsf((float)e[(((size_t)(b*23+s))*36 + w)*64 + d]));
  eT[((size_t)b*36+w)*64 + d] = mx;
}

// ---------------------------------------------------------------------------
// GAT layer (one workgroup per batch item)
// ---------------------------------------------------------------------------
struct GatW { const float *apB,*apW,*attw,*bnB,*bnS,*poB,*poW,*pwB,*pwW; };

__global__ void k_gat(const float* __restrict__ x, float* __restrict__ out,
                      GatW w, int N, int D, int O, float temp){
  int b = blockIdx.x, t = threadIdx.x;
  const int BT = 256;
  __shared__ float xs[36*64], Ab[36*36], axs[36*64];
  for(int p=t; p<N*D; p+=BT) xs[p] = x[(size_t)b*N*D + p];
  __syncthreads();
  for(int p=t; p<N*N; p+=BT){
    int n=p/N, m=p%N; float s=0.f;
    for(int o=0;o<O;++o){
      float a = w.apB[o];
      for(int d=0; d<D; ++d) a += xs[n*D+d]*xs[m*D+d]*w.apW[d*O+o];
      s += w.attw[o]*tanhf(a);
    }
    Ab[p]=s;
  }
  __syncthreads();
  for(int n=t; n<N; n+=BT){
    float mx=-3.0e38f;
    for(int m=0;m<N;++m) mx=fmaxf(mx,Ab[n*N+m]);
    float ss=0.f;
    for(int m=0;m<N;++m){ float e=expf((Ab[n*N+m]-mx)/temp); Ab[n*N+m]=e; ss+=e; }
    float inv=1.f/ss;
    for(int m=0;m<N;++m) Ab[n*N+m]*=inv;
  }
  __syncthreads();
  for(int p=t; p<N*D; p+=BT){
    int n=p/D, d=p%D; float s=0.f;
    for(int m=0;m<N;++m) s += Ab[n*N+m]*xs[m*D+d];
    axs[p]=s;
  }
  __syncthreads();
  for(int p=t; p<N*O; p+=BT){
    int n=p/O, o=p%O;
    float s = w.pwB[o] + w.poB[o];
    for(int d=0; d<D; ++d) s += axs[n*D+d]*w.pwW[d*O+o] + xs[n*D+d]*w.poW[d*O+o];
    out[(size_t)b*N*O + p] = selu_f(s*w.bnS[o]+w.bnB[o]);
  }
}

// ---------------------------------------------------------------------------
// graph pool: sigmoid score, top-k select (stable), gather h*score
// ---------------------------------------------------------------------------
__global__ void k_gpool(const float* __restrict__ h, const float* __restrict__ pw,
                        const float* __restrict__ pb, float* __restrict__ out,
                        int N, int D, int K){
  int b = blockIdx.x, t = threadIdx.x;
  __shared__ float sc[40]; __shared__ int idx[32]; __shared__ int used[40];
  for(int n=t; n<N; n+=blockDim.x){
    float s = pb[0];
    for(int d=0; d<D; ++d) s += h[((size_t)b*N+n)*D+d]*pw[d];
    sc[n] = 1.f/(1.f+expf(-s)); used[n]=0;
  }
  __syncthreads();
  if(t==0){
    for(int i=0;i<K;++i){
      float best=-1.f; int bi=0;
      for(int n=0;n<N;++n) if(!used[n] && sc[n]>best){ best=sc[n]; bi=n; }
      used[bi]=1; idx[i]=bi;
    }
  }
  __syncthreads();
  for(int p=t; p<K*D; p+=blockDim.x){
    int i=p/D, d=p%D, n=idx[i];
    out[((size_t)b*K+i)*D+d] = h[((size_t)b*N+n)*D+d]*sc[n];
  }
}

// ---------------------------------------------------------------------------
// heterogeneous graph attention (one workgroup per batch item)
// ---------------------------------------------------------------------------
struct HtrgW {
  const float *apB,*apW,*apMB,*apMW,*bnB,*bnS,*poB,*poW,*poMB,*poMW,
              *pt1B,*pt1W,*pt2B,*pt2W,*pwB,*pwW,*pwMB,*pwMW,*w11,*w12,*w22,*wM;
};

__global__ void k_htrg(const float* __restrict__ x1, const float* __restrict__ x2,
                       const float* __restrict__ mi, float* __restrict__ h1,
                       float* __restrict__ h2, float* __restrict__ mo,
                       HtrgW w, int n1, int n2, int Di, int O, float temp){
  int b = blockIdx.x, t = threadIdx.x;
  const int BT = 256;
  int N = n1+n2;
  __shared__ float xs[36*64], Ab[36*36], axs[36*64], sM[40], mv[64], matt[64];
  for(int d=t; d<Di; d+=BT) mv[d]=mi[b*Di+d];
  for(int p=t; p<N*Di; p+=BT){
    int n=p/Di, d=p%Di;
    const float *xin, *W, *Bb;
    if(n<n1){ xin=x1+((size_t)b*n1+n)*Di; W=w.pt1W; Bb=w.pt1B; }
    else    { xin=x2+((size_t)b*n2+(n-n1))*Di; W=w.pt2W; Bb=w.pt2B; }
    float s=Bb[d];
    for(int k=0;k<Di;++k) s += xin[k]*W[k*Di+d];
    xs[p]=s;
  }
  __syncthreads();
  for(int p=t; p<N*N; p+=BT){
    int n=p/N, m=p%N;
    const float* ws = (n<n1) ? (m<n1 ? w.w11 : w.w12) : (m<n1 ? w.w12 : w.w22);
    float s=0.f;
    for(int o=0;o<O;++o){
      float a=w.apB[o];
      for(int d=0; d<Di; ++d) a += xs[n*Di+d]*xs[m*Di+d]*w.apW[d*O+o];
      s += ws[o]*tanhf(a);
    }
    Ab[p]=s;
  }
  __syncthreads();
  for(int n=t; n<N; n+=BT){
    float mx=-3.0e38f;
    for(int m=0;m<N;++m) mx=fmaxf(mx,Ab[n*N+m]);
    float ss=0.f;
    for(int m=0;m<N;++m){ float e=expf((Ab[n*N+m]-mx)/temp); Ab[n*N+m]=e; ss+=e; }
    float inv=1.f/ss;
    for(int m=0;m<N;++m) Ab[n*N+m]*=inv;
  }
  for(int n=t; n<N; n+=BT){
    float s=0.f;
    for(int o=0;o<O;++o){
      float a=w.apMB[o];
      for(int d=0; d<Di; ++d) a += xs[n*Di+d]*mv[d]*w.apMW[d*O+o];
      s += w.wM[o]*tanhf(a);
    }
    sM[n]=s;
  }
  __syncthreads();
  if(t==0){
    float mx=-3.0e38f;
    for(int n=0;n<N;++n) mx=fmaxf(mx,sM[n]);
    float ss=0.f;
    for(int n=0;n<N;++n){ float e=expf((sM[n]-mx)/temp); sM[n]=e; ss+=e; }
    float inv=1.f/ss;
    for(int n=0;n<N;++n) sM[n]*=inv;
  }
  __syncthreads();
  for(int d=t; d<Di; d+=BT){
    float s=0.f;
    for(int n=0;n<N;++n) s += sM[n]*xs[n*Di+d];
    matt[d]=s;
  }
  __syncthreads();
  for(int o=t; o<O; o+=BT){
    float s = w.pwMB[o]+w.poMB[o];
    for(int d=0; d<Di; ++d) s += matt[d]*w.pwMW[d*O+o] + mv[d]*w.poMW[d*O+o];
    mo[b*O+o]=s;
  }
  for(int p=t; p<N*Di; p+=BT){
    int n=p/Di, d=p%Di; float s=0.f;
    for(int m=0;m<N;++m) s += Ab[n*N+m]*xs[m*Di+d];
    axs[p]=s;
  }
  __syncthreads();
  for(int p=t; p<N*O; p+=BT){
    int n=p/O, o=p%O;
    float s = w.pwB[o]+w.poB[o];
    for(int d=0; d<Di; ++d) s += axs[n*Di+d]*w.pwW[d*O+o] + xs[n*Di+d]*w.poW[d*O+o];
    s = selu_f(s*w.bnS[o]+w.bnB[o]);
    if(n<n1) h1[((size_t)b*n1+n)*O+o]=s;
    else     h2[((size_t)b*n2+(n-n1))*O+o]=s;
  }
}

// ---------------------------------------------------------------------------
// final readout: stats + linear(160->2) + softmax
// ---------------------------------------------------------------------------
__global__ void k_readout(const float* __restrict__ oT, const float* __restrict__ oS,
                          const float* __restrict__ mm, const float* __restrict__ ow,
                          const float* __restrict__ ob, float* __restrict__ dout){
  int b = blockIdx.x, t = threadIdx.x;
  __shared__ float last[160];
  if(t<32){
    float ma=0.f, me=0.f;
    for(int n=0;n<12;++n){ float v=oT[((size_t)b*12+n)*32+t]; ma=fmaxf(ma,fabsf(v)); me+=v; }
    last[t]=ma; last[32+t]=me/12.f;
    ma=0.f; me=0.f;
    for(int n=0;n<5;++n){ float v=oS[((size_t)b*5+n)*32+t]; ma=fmaxf(ma,fabsf(v)); me+=v; }
    last[64+t]=ma; last[96+t]=me/5.f;
    last[128+t]=mm[b*32+t];
  }
  __syncthreads();
  if(t==0){
    float l0=ob[0], l1=ob[1];
    for(int i=0;i<160;++i){ l0+=last[i]*ow[i*2]; l1+=last[i]*ow[i*2+1]; }
    float m=fmaxf(l0,l1);
    float e0=expf(l0-m), e1=expf(l1-m), s=e0+e1;
    dout[b*2]=e0/s; dout[b*2+1]=e1/s;
  }
}

// ===========================================================================
// host orchestration
// ===========================================================================
// workspace byte offsets (all 256-aligned)
static constexpr size_t O_XH     = 0;                       // x f16 2,560,000
static constexpr size_t O_WSINC  = 2560000;                 // 80x128 f16
static constexpr size_t O_WC2_0  = 2580480;
static constexpr size_t O_WC1_1  = 2592768;
static constexpr size_t O_WC2_1  = 2605056;
static constexpr size_t O_WC1_2  = 2617344;
static constexpr size_t O_WC2_2  = 2641920;
static constexpr size_t O_WDN_2  = 2691072;
static constexpr size_t O_WC1_3  = 2703360;
static constexpr size_t O_WC2_3  = 2752512;
static constexpr size_t O_WC1_4  = 2801664;
static constexpr size_t O_WC2_4  = 2850816;
static constexpr size_t O_WC1_5  = 2899968;
static constexpr size_t O_WC2_5  = 2949120;
static constexpr size_t O_ACT0   = 2998272;                 // (16,23,26624) f16
static constexpr size_t O_Y1     = 22593536;                // y1 region (max 654MB, blk0)
static constexpr size_t O_BN     = O_Y1 + 218087424;        // bn-act region (blk>=1; temporal reuse of blk0 y1 tail)
static constexpr size_t O_ACTA   = 676708352;               // ping (max 209MB)
static constexpr size_t O_ACTB   = 885708800;               // pong (max 139MB)
static constexpr size_t G0       = 1025042432;              // graph f32 region
static constexpr size_t g_eS  = G0,            g_eT  = g_eS +94208,  g_gS  = g_eT +147456,
                        g_gT  = g_gS +94208,   g_outS= g_gT +147456, g_outT= g_outS+45056,
                        g_m1a = g_outT+102400, g_m2a = g_m1a +4096,  g_m1b = g_m2a +4096,
                        g_m2b = g_m1b +2048,   g_hT1 = g_m2b +2048,  g_hS1 = g_hT1 +51200,
                        g_hT2 = g_hS1 +22528,  g_hS2 = g_hT2 +51200, g_oT1 = g_hS2 +22528,
                        g_oS1 = g_oT1 +24576,  g_oT2 = g_oS1 +10240, g_oS2 = g_oT2 +24576,
                        g_tA  = g_oS2 +10240,  g_sA  = g_tA  +24576, g_mA  = g_sA  +10240,
                        g_tA2 = g_mA  +2048,   g_sA2 = g_tA2 +24576, g_mA2 = g_sA2 +10240,
                        g_oT  = g_mA2 +2048,   g_oS  = g_oT  +24576, g_mm  = g_oS  +10240,
                        G_END = g_mm  +2048;   // ~0.98 GiB total

static void pack_w(const float* w, f16* o, int CO,int CI,int KH,int KW,int COp,int Kp,
                   hipStream_t s){
  int tot = COp*Kp;
  k_pack_conv<<<(tot+255)/256, 256, 0, s>>>(w, o, CO, CI, KH, KW, COp, Kp);
}

static GatW mk_gat(const float* const* IN, int base){
  GatW g;
  g.apB=IN[base]; g.apW=IN[base+1]; g.attw=IN[base+2]; g.bnB=IN[base+3];
  g.bnS=IN[base+4]; g.poB=IN[base+5]; g.poW=IN[base+6]; g.pwB=IN[base+7];
  g.pwW=IN[base+8];
  return g;
}
static HtrgW mk_htrg(const float* const* IN, int base){
  HtrgW w;
  w.apB=IN[base];    w.apW=IN[base+1];  w.apMB=IN[base+2]; w.apMW=IN[base+3];
  w.bnB=IN[base+4];  w.bnS=IN[base+5];  w.poB=IN[base+6];  w.poW=IN[base+7];
  w.poMB=IN[base+8]; w.poMW=IN[base+9]; w.pt1B=IN[base+10];w.pt1W=IN[base+11];
  w.pt2B=IN[base+12];w.pt2W=IN[base+13];w.pwB=IN[base+14]; w.pwW=IN[base+15];
  w.pwMB=IN[base+16];w.pwMW=IN[base+17];w.w11=IN[base+18]; w.w12=IN[base+19];
  w.w22=IN[base+20]; w.wM=IN[base+21];
  return w;
}

extern "C" void kernel_launch(void* const* d_in, const int* in_sizes, int n_in,
                              void* d_out, int out_size, void* d_ws, size_t ws_size,
                              hipStream_t stream){
  const float* const* IN = (const float* const*)d_in;
  char* ws = (char*)d_ws;
  if(ws_size < G_END) return;  // scratch too small; nothing sane to do
  auto W16 = [&](size_t o){ return (f16*)(ws+o); };
  auto F32 = [&](size_t o){ return (float*)(ws+o); };

  // ---- prep: x -> f16, pack weights -----------------------------------
  k_cvt<<<(1280000+255)/256, 256, 0, stream>>>(IN[176], W16(O_XH), 1280000);
  pack_w(IN[87], W16(O_WSINC), 70, 1, 1, 128, 80, 128, stream);
  pack_w(IN[ 5], W16(O_WC2_0), 32,32,2,3, 32,192, stream);
  pack_w(IN[13], W16(O_WC1_1), 32,32,2,3, 32,192, stream);
  pack_w(IN[15], W16(O_WC2_1), 32,32,2,3, 32,192, stream);
  pack_w(IN[21], W16(O_WC1_2), 64,32,2,3, 64,192, stream);
  pack_w(IN[23], W16(O_WC2_2), 64,64,2,3, 64,384, stream);
  pack_w(IN[25], W16(O_WDN_2), 64,32,1,3, 64, 96, stream);
  pack_w(IN[31], W16(O_WC1_3), 64,64,2,3, 64,384, stream);
  pack_w(IN[33], W16(O_WC2_3), 64,64,2,3, 64,384, stream);
  pack_w(IN[39], W16(O_WC1_4), 64,64,2,3, 64,384, stream);
  pack_w(IN[41], W16(O_WC2_4), 64,64,2,3, 64,384, stream);
  pack_w(IN[47], W16(O_WC1_5), 64,64,2,3, 64,384, stream);
  pack_w(IN[49], W16(O_WC2_5), 64,64,2,3, 64,384, stream);

  // ---- sinc conv + |.| + pool(3,3) + bn + selu (WMMA) -----------------
  k_sincconv<<<dim3(1664,16), 32, 0, stream>>>(W16(O_XH), W16(O_WSINC),
                                               IN[51], IN[50], W16(O_ACT0));

  // ---- encoder --------------------------------------------------------
  // block0: 1->32, W 26624 -> 8874
  k_conv1_c1<<<dim3((NB_W0+255)/256,24,16), 256, 0, stream>>>(
      W16(O_ACT0), IN[3], IN[2], IN[1], IN[0], W16(O_Y1));
  k_convB<<<dim3(555,23,32), 32, 0, stream>>>(
      W16(O_Y1), W16(O_WC2_0), IN[4], W16(O_ACT0),
      nullptr, nullptr, IN[7], IN[6], W16(O_ACTA),
      26624, 8874, 32, 1, 32, 2);

  // block1: 32->32, W 8874 -> 2958
  { long n = (long)16*23*8874*32;
    k_bnselu<<<(unsigned)((n+255)/256), 256, 0, stream>>>(W16(O_ACTA), W16(O_BN), IN[9], IN[8], n, 31);
    k_convA<<<dim3(555,24,32), 32, 0, stream>>>(W16(O_BN), W16(O_WC1_1), IN[12], IN[11], IN[10], W16(O_Y1), 8874, 32, 32);
    k_convB<<<dim3(185,23,32), 32, 0, stream>>>(W16(O_Y1), W16(O_WC2_1), IN[14], W16(O_ACTA),
      nullptr, nullptr, nullptr, nullptr, W16(O_ACTB), 8874, 2958, 32, 32, 32, 0); }

  // block2: 32->64 (down), W 2958 -> 986
  { long n = (long)16*23*2958*32;
    k_bnselu<<<(unsigned)((n+255)/256), 256, 0, stream>>>(W16(O_ACTB), W16(O_BN), IN[17], IN[16], n, 31);
    k_convA<<<dim3(185,24,64), 32, 0, stream>>>(W16(O_BN), W16(O_WC1_2), IN[20], IN[19], IN[18], W16(O_Y1), 2958, 32, 64);
    k_convB<<<dim3(62,23,64), 32, 0, stream>>>(W16(O_Y1), W16(O_WC2_2), IN[22], W16(O_ACTB),
      W16(O_WDN_2), IN[24], nullptr, nullptr, W16(O_ACTA), 2958, 986, 64, 32, 64, 1); }

  // block3: 64->64, W 986 -> 328
  { long n = (long)16*23*986*64;
    k_bnselu<<<(unsigned)((n+255)/256), 256, 0, stream>>>(W16(O_ACTA), W16(O_BN), IN[27], IN[26], n, 63);
    k_convA<<<dim3(62,24,64), 32, 0, stream>>>(W16(O_BN), W16(O_WC1_3), IN[30], IN[29], IN[28], W16(O_Y1), 986, 64, 64);
    k_convB<<<dim3(21,23,64), 32, 0, stream>>>(W16(O_Y1), W16(O_WC2_3), IN[32], W16(O_ACTA),
      nullptr, nullptr, nullptr, nullptr, W16(O_ACTB), 986, 328, 64, 64, 64, 0); }

  // block4: 64->64, W 328 -> 109
  { long n = (long)16*23*328*64;
    k_bnselu<<<(unsigned)((n+255)/256), 256, 0, stream>>>(W16(O_ACTB), W16(O_BN), IN[35], IN[34], n, 63);
    k_convA<<<dim3(21,24,64), 32, 0, stream>>>(W16(O_BN), W16(O_WC1_4), IN[38], IN[37], IN[36], W16(O_Y1), 328, 64, 64);
    k_convB<<<dim3(7,23,64), 32, 0, stream>>>(W16(O_Y1), W16(O_WC2_4), IN[40], W16(O_ACTB),
      nullptr, nullptr, nullptr, nullptr, W16(O_ACTA), 328, 109, 64, 64, 64, 0); }

  // block5: 64->64, W 109 -> 36   (e = ACTB = (16,23,36,64) f16)
  { long n = (long)16*23*109*64;
    k_bnselu<<<(unsigned)((n+255)/256), 256, 0, stream>>>(W16(O_ACTA), W16(O_BN), IN[43], IN[42], n, 63);
    k_convA<<<dim3(7,24,64), 32, 0, stream>>>(W16(O_BN), W16(O_WC1_5), IN[46], IN[45], IN[44], W16(O_Y1), 109, 64, 64);
    k_convB<<<dim3(3,23,64), 32, 0, stream>>>(W16(O_Y1), W16(O_WC2_5), IN[48], W16(O_ACTA),
      nullptr, nullptr, nullptr, nullptr, W16(O_ACTB), 109, 36, 64, 64, 64, 0); }

  // ---- graph attention -------------------------------------------------
  k_eS<<<dim3(23,16), 64, 0, stream>>>(W16(O_ACTB), IN[86], F32(g_eS));
  k_eT<<<dim3(36,16), 64, 0, stream>>>(W16(O_ACTB), F32(g_eT));

  k_gat<<<16,256,0,stream>>>(F32(g_eS), F32(g_gS), mk_gat(IN,52), 23,64,64, 2.0f);
  k_gat<<<16,256,0,stream>>>(F32(g_eT), F32(g_gT), mk_gat(IN,61), 36,64,64, 2.0f);
  k_gpool<<<16,256,0,stream>>>(F32(g_gS), IN[75], IN[74], F32(g_outS), 23,64,11);
  k_gpool<<<16,256,0,stream>>>(F32(g_gT), IN[77], IN[76], F32(g_outT), 36,64,25);

  k_bcast<<<16,64,0,stream>>>(IN[70], F32(g_m1a), 64);
  k_bcast<<<16,64,0,stream>>>(IN[71], F32(g_m2a), 64);

  // branch 1: st11 -> pools -> st12 -> residual add
  k_htrg<<<16,256,0,stream>>>(F32(g_outT), F32(g_outS), F32(g_m1a),
                              F32(g_hT1), F32(g_hS1), F32(g_m1b),
                              mk_htrg(IN,88), 25,11,64,32, 100.f);
  k_gpool<<<16,256,0,stream>>>(F32(g_hS1), IN[79], IN[78], F32(g_oS1), 11,32,5);
  k_gpool<<<16,256,0,stream>>>(F32(g_hT1), IN[83], IN[82], F32(g_oT1), 25,32,12);
  k_htrg<<<16,256,0,stream>>>(F32(g_oT1), F32(g_oS1), F32(g_m1b),
                              F32(g_tA), F32(g_sA), F32(g_mA),
                              mk_htrg(IN,110), 12,5,32,32, 100.f);
  k_add<<<(6144+255)/256,256,0,stream>>>(F32(g_oT1), F32(g_tA), 6144);
  k_add<<<(2560+255)/256,256,0,stream>>>(F32(g_oS1), F32(g_sA), 2560);
  k_add<<<2,256,0,stream>>>(F32(g_m1b), F32(g_mA), 512);

  // branch 2
  k_htrg<<<16,256,0,stream>>>(F32(g_outT), F32(g_outS), F32(g_m2a),
                              F32(g_hT2), F32(g_hS2), F32(g_m2b),
                              mk_htrg(IN,132), 25,11,64,32, 100.f);
  k_gpool<<<16,256,0,stream>>>(F32(g_hS2), IN[81], IN[80], F32(g_oS2), 11,32,5);
  k_gpool<<<16,256,0,stream>>>(F32(g_hT2), IN[85], IN[84], F32(g_oT2), 25,32,12);
  k_htrg<<<16,256,0,stream>>>(F32(g_oT2), F32(g_oS2), F32(g_m2b),
                              F32(g_tA2), F32(g_sA2), F32(g_mA2),
                              mk_htrg(IN,154), 12,5,32,32, 100.f);
  k_add<<<(6144+255)/256,256,0,stream>>>(F32(g_oT2), F32(g_tA2), 6144);
  k_add<<<(2560+255)/256,256,0,stream>>>(F32(g_oS2), F32(g_sA2), 2560);
  k_add<<<2,256,0,stream>>>(F32(g_m2b), F32(g_mA2), 512);

  // merge + readout
  k_maxel<<<(6144+255)/256,256,0,stream>>>(F32(g_oT1), F32(g_oT2), F32(g_oT), 6144);
  k_maxel<<<(2560+255)/256,256,0,stream>>>(F32(g_oS1), F32(g_oS2), F32(g_oS), 2560);
  k_maxel<<<2,256,0,stream>>>(F32(g_m1b), F32(g_m2b), F32(g_mm), 512);
  k_readout<<<16,64,0,stream>>>(F32(g_oT), F32(g_oS), F32(g_mm),
                                IN[73], IN[72], (float*)d_out);
}
